// CombinedModel_52312701665788
// MI455X (gfx1250) — compile-verified, matrix operand order checked by reading, not run on hardware
//
#include <hip/hip_runtime.h>
#include <hip/hip_bf16.h>
#include <math.h>

typedef __attribute__((ext_vector_type(16))) _Float16 v16h;
typedef __attribute__((ext_vector_type(8)))  _Float16 v8h;
typedef __attribute__((ext_vector_type(8)))  float    v8f;

#define NPRED     25200
#define MAX_DET   300
#define MPAD      304          // 19 * 16
#define IMG       640
#define CROP      64
#define FLATK     16384        // 16 * 32 * 32
#define NHEAD     8
#define DHID      128
#define CONF_T    0.25f
#define IOU_T     0.45f
#define PROV      38
#define ALPHA     25
#define AD        35
#define NCOLS     (PROV + ALPHA + 6*AD)   // 273

// ---------------------------------------------------------------------------
// 1) decode: score = obj*cls, boxes = xywh -> xyxy, masked score
// ---------------------------------------------------------------------------
__global__ __launch_bounds__(256) void decode_kernel(const float* __restrict__ preds,
                                                     float* __restrict__ boxes,
                                                     float* __restrict__ scores)
{
    for (int i = blockIdx.x * blockDim.x + threadIdx.x; i < NPRED;
         i += gridDim.x * blockDim.x) {
        const float* p = preds + (size_t)i * 6;
        float s = p[4] * p[5];
        scores[i] = (s > CONF_T) ? s : -INFINITY;
        float hx = p[2] * 0.5f, hy = p[3] * 0.5f;
        boxes[i*4 + 0] = p[0] - hx;
        boxes[i*4 + 1] = p[1] - hy;
        boxes[i*4 + 2] = p[0] + hx;
        boxes[i*4 + 3] = p[1] + hy;
    }
}

// ---------------------------------------------------------------------------
// 2) top-k 300 by iterative argmax (single workgroup, tie -> lowest index,
//    matching lax.top_k ordering)
// ---------------------------------------------------------------------------
__global__ __launch_bounds__(1024) void topk_kernel(float* __restrict__ scores,
                                                    int* __restrict__ tidx,
                                                    float* __restrict__ tval)
{
    __shared__ float sv[1024];
    __shared__ int   si[1024];
    const int tid = threadIdx.x;
    for (int t = 0; t < MAX_DET; ++t) {
        float best = -INFINITY; int bi = 0x7fffffff;
        for (int i = tid; i < NPRED; i += 1024) {
            float v = scores[i];
            if (v > best || (v == best && i < bi)) { best = v; bi = i; }
        }
        sv[tid] = best; si[tid] = bi;
        __syncthreads();
        for (int s = 512; s > 0; s >>= 1) {
            if (tid < s) {
                float v2 = sv[tid + s]; int i2 = si[tid + s];
                if (v2 > sv[tid] || (v2 == sv[tid] && i2 < si[tid])) {
                    sv[tid] = v2; si[tid] = i2;
                }
            }
            __syncthreads();
        }
        if (tid == 0) {
            tidx[t] = si[0];
            tval[t] = sv[0];
            scores[si[0]] = -INFINITY;
        }
        __syncthreads();
    }
}

// ---------------------------------------------------------------------------
// 3) NMS (sequential suppression, one workgroup) + crop coords
// ---------------------------------------------------------------------------
__global__ __launch_bounds__(320) void nms_kernel(const float* __restrict__ boxes,
                                                  const int* __restrict__ tidx,
                                                  const float* __restrict__ tval,
                                                  int* __restrict__ keep,
                                                  int* __restrict__ cx,
                                                  int* __restrict__ cy)
{
    __shared__ float bx[MAX_DET][4];
    __shared__ int   kp[MAX_DET];
    __shared__ int   flag;
    const int j = threadIdx.x;
    if (j < MAX_DET) {
        int id = tidx[j];
        bx[j][0] = boxes[id*4+0]; bx[j][1] = boxes[id*4+1];
        bx[j][2] = boxes[id*4+2]; bx[j][3] = boxes[id*4+3];
        kp[j] = (tval[j] > CONF_T) ? 1 : 0;
        int xx = (int)rintf(bx[j][0]);               // round-half-even == jnp.round
        int yy = (int)rintf(bx[j][1]);
        xx = xx < 0 ? 0 : (xx > IMG - CROP ? IMG - CROP : xx);
        yy = yy < 0 ? 0 : (yy > IMG - CROP ? IMG - CROP : yy);
        cx[j] = xx; cy[j] = yy;
    }
    __syncthreads();
    for (int i = 1; i < MAX_DET; ++i) {
        if (j == 0) flag = 0;
        __syncthreads();
        if (j < i && kp[j]) {
            float ax1 = bx[i][0], ay1 = bx[i][1], ax2 = bx[i][2], ay2 = bx[i][3];
            float bx1 = bx[j][0], by1 = bx[j][1], bx2 = bx[j][2], by2 = bx[j][3];
            float areaA = (ax2 - ax1) * (ay2 - ay1);
            float areaB = (bx2 - bx1) * (by2 - by1);
            float iw = fminf(ax2, bx2) - fmaxf(ax1, bx1); iw = fmaxf(iw, 0.f);
            float ih = fminf(ay2, by2) - fmaxf(ay1, by1); ih = fmaxf(ih, 0.f);
            float inter = iw * ih;
            float iou = inter / (areaA + areaB - inter + 1e-7f);
            if (iou > IOU_T) flag = 1;               // benign race, all write 1
        }
        __syncthreads();
        if (j == 0) kp[i] = kp[i] && !flag;
        __syncthreads();
    }
    if (j < MAX_DET) keep[j] = kp[j];
}

// ---------------------------------------------------------------------------
// 4) crop + 3x3 conv(16) SAME + ReLU + 2x2 maxpool -> feats row (f16)
//    one block per detection; rows 300..303 zero-padded for the GEMM
// ---------------------------------------------------------------------------
__global__ __launch_bounds__(256) void crop_conv_kernel(const float* __restrict__ image,
                                                        const float* __restrict__ Wc,
                                                        const float* __restrict__ bc,
                                                        const int* __restrict__ cx,
                                                        const int* __restrict__ cy,
                                                        _Float16* __restrict__ feats)
{
    const int k   = blockIdx.x;
    const int tid = threadIdx.x;
    _Float16* frow = feats + (size_t)k * FLATK;
    if (k >= MAX_DET) {                       // pad rows -> zeros
        for (int i = tid; i < FLATK; i += 256) frow[i] = (_Float16)0.f;
        return;
    }
    __shared__ float simg[3][66][66];         // zero-padded crop, ~52 KB
    __shared__ float sw[16][27];
    __shared__ float sb[16];
    for (int i = tid; i < 3*66*66; i += 256) ((float*)simg)[i] = 0.f;
    for (int i = tid; i < 16*27;   i += 256) ((float*)sw)[i]  = Wc[i];
    if (tid < 16) sb[tid] = bc[tid];
    __syncthreads();
    const int x0 = cx[k], y0 = cy[k];
    for (int i = tid; i < 3*CROP*CROP; i += 256) {
        int c = i >> 12;                 // /4096
        int r = (i >> 6) & 63;
        int x = i & 63;
        simg[c][r+1][x+1] = image[(size_t)c*IMG*IMG + (size_t)(y0 + r)*IMG + (x0 + x)];
    }
    __syncthreads();
    for (int po = tid; po < FLATK; po += 256) {
        int oc  = po >> 10;
        int rem = po & 1023;
        int py  = rem >> 5;
        int px  = rem & 31;
        float m = -INFINITY;
        #pragma unroll
        for (int dy = 0; dy < 2; ++dy)
        #pragma unroll
        for (int dx = 0; dx < 2; ++dx) {
            int y = 2*py + dy, x = 2*px + dx;
            float acc = sb[oc];
            #pragma unroll
            for (int ci = 0; ci < 3; ++ci)
            #pragma unroll
            for (int ky = 0; ky < 3; ++ky)
            #pragma unroll
            for (int kx = 0; kx < 3; ++kx)
                acc += simg[ci][y + ky][x + kx] * sw[oc][ci*9 + ky*3 + kx];
            acc = fmaxf(acc, 0.f);        // ReLU before maxpool (commutes anyway)
            m = fmaxf(m, acc);
        }
        frow[po] = (_Float16)m;
    }
}

// ---------------------------------------------------------------------------
// 5) pack W1 (fp32, [8][16384][128]) into per-lane WMMA B-fragment layout, f16
//    packed[((h*512+ks)*8+nt)*32+lane][16 halves contiguous]
//    lane<16 : n=lane,     K = ks*32 + {0..7, 16..23}
//    lane>=16: n=lane-16,  K = ks*32 + {8..15, 24..31}
// ---------------------------------------------------------------------------
__global__ __launch_bounds__(256) void pack_w1_kernel(const float* __restrict__ W1,
                                                      _Float16* __restrict__ w1p)
{
    const size_t total = (size_t)NHEAD * FLATK * DHID;   // 2^24
    for (size_t pidx = (size_t)blockIdx.x * blockDim.x + threadIdx.x; pidx < total;
         pidx += (size_t)gridDim.x * blockDim.x) {
        int j    = (int)(pidx & 15);
        int lane = (int)((pidx >> 4) & 31);
        int nt   = (int)((pidx >> 9) & 7);
        int ks   = (int)((pidx >> 12) & 511);
        int h    = (int)(pidx >> 21);
        int n    = lane & 15;
        int kofs = (lane < 16) ? 0 : 8;
        int kk   = ks*32 + kofs + ((j < 8) ? j : j + 8);
        int col  = nt*16 + n;
        w1p[pidx] = (_Float16)W1[((size_t)h * FLATK + kk) * DHID + col];
    }
}

// ---------------------------------------------------------------------------
// 6) the big GEMM: h[head] = relu(feats(304x16384) @ W1[head](16384x128) + b1)
//    block = 256 thr = 8 waves; wave w owns N-tile w; grid = (19 M-tiles, 8 heads)
//    A staged in LDS (16 x 256 halves), B from packed fragments (L2-resident)
// ---------------------------------------------------------------------------
__global__ __launch_bounds__(256) void gemm_kernel(const _Float16* __restrict__ feats,
                                                   const _Float16* __restrict__ w1p,
                                                   const float* __restrict__ b1,
                                                   float* __restrict__ hbuf)
{
    const int mtile = blockIdx.x;     // 0..18
    const int head  = blockIdx.y;     // 0..7
    const int tid   = threadIdx.x;
    const int wave  = tid >> 5;       // N-tile
    const int lane  = tid & 31;

    __shared__ _Float16 As[16][256];  // 8 KB

    const _Float16* Abase = feats + (size_t)mtile * 16 * FLATK;
    const int m    = lane & 15;
    const int kofs = (lane < 16) ? 0 : 8;
    const int r      = tid >> 4;          // cooperative-load row  (0..15)
    const int cchunk = (tid & 15) * 16;   // cooperative-load cols (16 halves)

    v8f acc = {};
    for (int kb = 0; kb < FLATK; kb += 256) {
        __syncthreads();                  // WAR guard on As
        const _Float16* src = Abase + (size_t)r * FLATK + kb + cchunk;
        *(float4*)&As[r][cchunk]     = *(const float4*)src;        // 8 halves
        *(float4*)&As[r][cchunk + 8] = *(const float4*)(src + 8);  // 8 halves
        __builtin_prefetch(src + 256, 0, 1);                       // next A block
        __syncthreads();
        #pragma unroll
        for (int kk = 0; kk < 256; kk += 32) {
            v8h lo = *(const v8h*)&As[m][kk + kofs];        // K +0..7   (or +8..15)
            v8h hi = *(const v8h*)&As[m][kk + kofs + 16];   // K +16..23 (or +24..31)
            v16h a;
            #pragma unroll
            for (int q = 0; q < 8; ++q) { a[q] = lo[q]; a[q + 8] = hi[q]; }
            const int ks = (kb + kk) >> 5;
            const v16h* bp = (const v16h*)(w1p +
                ((((size_t)head * 512 + ks) * 8 + wave) * 32 + lane) * 16);
            v16h b = *bp;
            acc = __builtin_amdgcn_wmma_f32_16x16x32_f16(
                      false, a, false, b, (short)0, acc, false, false);
        }
    }
    // epilogue: C layout VGPR i -> M = i + (lane<16 ? 0 : 8), N = lane&15
    const int n = lane & 15;
    const int d = wave * 16 + n;
    const float bias  = b1[head * DHID + d];
    const int   mbase = mtile * 16 + ((lane < 16) ? 0 : 8);
    #pragma unroll
    for (int i = 0; i < 8; ++i) {
        float v = acc[i] + bias;
        v = v > 0.f ? v : 0.f;
        hbuf[((size_t)head * MPAD + (mbase + i)) * DHID + d] = v;
    }
}

// ---------------------------------------------------------------------------
// 7) head projections + keep mask -> logits (300 x 273)
// ---------------------------------------------------------------------------
__global__ __launch_bounds__(288) void head_kernel(const float* __restrict__ hbuf,
                                                   const float* __restrict__ W2p,
                                                   const float* __restrict__ b2p,
                                                   const float* __restrict__ W2a,
                                                   const float* __restrict__ b2a,
                                                   const float* __restrict__ W2d,
                                                   const float* __restrict__ b2d,
                                                   const int* __restrict__ keep,
                                                   float* __restrict__ out)
{
    const int k = blockIdx.x;
    const int c = threadIdx.x;
    if (c >= NCOLS) return;
    const float* hrow; const float* w; float bias; int ldw;
    if (c < PROV) {
        hrow = hbuf + ((size_t)0 * MPAD + k) * DHID;
        w = W2p + c; ldw = PROV; bias = b2p[c];
    } else if (c < PROV + ALPHA) {
        int o = c - PROV;
        hrow = hbuf + ((size_t)1 * MPAD + k) * DHID;
        w = W2a + o; ldw = ALPHA; bias = b2a[o];
    } else {
        int jj = c - (PROV + ALPHA);
        int h2 = jj / AD, o = jj % AD;
        hrow = hbuf + ((size_t)(2 + h2) * MPAD + k) * DHID;
        w = W2d + (size_t)h2 * DHID * AD + o; ldw = AD; bias = b2d[h2 * AD + o];
    }
    float acc = bias;
    #pragma unroll 4
    for (int d = 0; d < DHID; ++d) acc += hrow[d] * w[(size_t)d * ldw];
    out[(size_t)k * NCOLS + c] = keep[k] ? acc : 0.f;
}

// ---------------------------------------------------------------------------
extern "C" void kernel_launch(void* const* d_in, const int* in_sizes, int n_in,
                              void* d_out, int out_size, void* d_ws, size_t ws_size,
                              hipStream_t stream)
{
    (void)in_sizes; (void)n_in; (void)out_size; (void)ws_size;
    const float* preds  = (const float*)d_in[0];
    const float* image  = (const float*)d_in[1];
    const float* W_conv = (const float*)d_in[2];
    const float* b_conv = (const float*)d_in[3];
    const float* W1     = (const float*)d_in[4];
    const float* b1     = (const float*)d_in[5];
    const float* W2p    = (const float*)d_in[6];
    const float* b2p    = (const float*)d_in[7];
    const float* W2a    = (const float*)d_in[8];
    const float* b2a    = (const float*)d_in[9];
    const float* W2d    = (const float*)d_in[10];
    const float* b2d    = (const float*)d_in[11];
    float* out = (float*)d_out;

    char* ws = (char*)d_ws;
    size_t off = 0;
    auto alloc = [&](size_t bytes) -> char* {
        size_t o = (off + 255) & ~(size_t)255;
        off = o + bytes;
        return ws + o;
    };
    float*    boxes  = (float*)   alloc((size_t)NPRED * 4 * sizeof(float));
    float*    scores = (float*)   alloc((size_t)NPRED * sizeof(float));
    int*      tidx   = (int*)     alloc(MAX_DET * sizeof(int));
    float*    tval   = (float*)   alloc(MAX_DET * sizeof(float));
    int*      keep   = (int*)     alloc(MAX_DET * sizeof(int));
    int*      cx     = (int*)     alloc(MAX_DET * sizeof(int));
    int*      cy     = (int*)     alloc(MAX_DET * sizeof(int));
    float*    hbuf   = (float*)   alloc((size_t)NHEAD * MPAD * DHID * sizeof(float));
    _Float16* feats  = (_Float16*)alloc((size_t)MPAD * FLATK * sizeof(_Float16));
    _Float16* w1p    = (_Float16*)alloc((size_t)NHEAD * FLATK * DHID * sizeof(_Float16));

    decode_kernel   <<<(NPRED + 255) / 256, 256, 0, stream>>>(preds, boxes, scores);
    pack_w1_kernel  <<<8192, 256, 0, stream>>>(W1, w1p);
    topk_kernel     <<<1, 1024, 0, stream>>>(scores, tidx, tval);
    nms_kernel      <<<1, 320, 0, stream>>>(boxes, tidx, tval, keep, cx, cy);
    crop_conv_kernel<<<MPAD, 256, 0, stream>>>(image, W_conv, b_conv, cx, cy, feats);
    gemm_kernel     <<<dim3(MPAD / 16, NHEAD), 256, 0, stream>>>(feats, w1p, b1, hbuf);
    head_kernel     <<<MAX_DET, 288, 0, stream>>>(hbuf, W2p, b2p, W2a, b2a, W2d, b2d,
                                                  keep, out);
}
